// NodesLayer_9517647527940
// MI455X (gfx1250) — compile-verified, matrix operand order checked by reading, not run on hardware
//
#include <hip/hip_runtime.h>

// ---- CDNA5 vector types (avoid HIP class-type vectors inside bit_cast) ----
typedef __attribute__((ext_vector_type(16))) __bf16     v16bf;
typedef __attribute__((ext_vector_type(2)))  __bf16     v2bf;
typedef __attribute__((ext_vector_type(8)))  float      v8f;
typedef __attribute__((ext_vector_type(4)))  float      f32x4;
typedef __attribute__((ext_vector_type(4)))  unsigned   u32x4;
typedef __attribute__((ext_vector_type(2)))  unsigned   u32x2;

#define B_   32
#define N_   1024
#define M_   256
#define H_   128
#define BLKR 128          // output rows per workgroup
#define KC   32           // K per bf16 WMMA

// ---- LDS layout (bytes). Everything 16B aligned. Total 203,264 < 320KB. ----
#define LDA   40          // As  k-stride (ushorts), 128 rows  -> 10,240 B
#define LDH   40          // hT  k-stride, 256 rows            -> 20,480 B
#define LDKB  264         // kb  k-stride, 128 rows            -> 67,584 B
#define LDW1  264         // W1T k-stride (K=M=256), 128 rows  -> 67,584 B
#define LDW2  136         // W2T k-stride (K=H=128), 256 rows  -> 69,632 B
#define LDTB  136         // tb  k-stride, 128 rows            -> 34,816 B
#define OFF_AS    0
#define OFF_HT    10240
#define OFF_KB    30720
#define OFF_WB    98304   // holds W1T first, then W2T (max 69,632 B)
#define OFF_TB    167936
#define OFF_MASK  202752  // 128 floats
#define SMEM_BYTES 203264

struct Frag { u32x4 q0, q1; };   // 32B = one v16bf WMMA operand

__device__ __forceinline__ unsigned short f2bf(float f) {
  unsigned u = __builtin_bit_cast(unsigned, f);
  unsigned r = (u + 0x7FFFu + ((u >> 16) & 1u)) >> 16;   // RNE
  return (unsigned short)r;
}

// Pack two floats into one dword of bf16x2 (hardware packed convert if present).
__device__ __forceinline__ unsigned pack_bf16x2(float lo, float hi) {
#if __has_builtin(__builtin_amdgcn_cvt_pk_bf16_f32)
  v2bf p = __builtin_amdgcn_cvt_pk_bf16_f32(lo, hi);
  return __builtin_bit_cast(unsigned, p);
#else
  return (unsigned)f2bf(lo) | ((unsigned)f2bf(hi) << 16);
#endif
}

// A/B operand load per ISA 7.12.2 (16-bit 16x32): lane-half hl picks K 0-7/8-15
// in elements 0..7 and K 16-23/24-31 in elements 8..15; rows k-contiguous in LDS.
__device__ __forceinline__ Frag load_frag(const unsigned short* base, int row,
                                          int ld, int kOff, int hl) {
  const char* p = (const char*)(base + row * ld + kOff + hl * 8);
  Frag f;
  f.q0 = *(const u32x4*)(p);
  f.q1 = *(const u32x4*)(p + 32);
  return f;
}

__device__ __forceinline__ v8f wmma_bf16(const Frag& a, const Frag& b, v8f c) {
  return __builtin_amdgcn_wmma_f32_16x16x32_bf16(
      false, __builtin_bit_cast(v16bf, a),
      false, __builtin_bit_cast(v16bf, b),
      (short)0, c, false, false);
}

__global__ void __launch_bounds__(256)
nodes_fused_kernel(const float* __restrict__ A,  const float* __restrict__ h,
                   const float* __restrict__ W1, const float* __restrict__ b1,
                   const float* __restrict__ W2, const float* __restrict__ b2,
                   float* __restrict__ out) {
  extern __shared__ char smem[];
  unsigned short* As = (unsigned short*)(smem + OFF_AS);
  unsigned short* hT = (unsigned short*)(smem + OFF_HT);
  unsigned short* kb = (unsigned short*)(smem + OFF_KB);
  unsigned short* Wb = (unsigned short*)(smem + OFF_WB);
  unsigned short* tb = (unsigned short*)(smem + OFF_TB);
  float* maskLds     = (float*)(smem + OFF_MASK);

  const int tid  = threadIdx.x;
  const int lane = tid & 31;
  const int wave = tid >> 5;          // 8 waves
  const int r    = lane & 15;
  const int hl   = lane >> 4;
  const int wr   = wave >> 2;         // 2 wave-rows  (64 rows each)
  const int wc   = wave & 3;          // 4 wave-cols  (64 cols each)

  const int blk   = blockIdx.x;       // 0..255
  const int b     = blk >> 3;
  const int mBase = (blk & 7) * BLKR;

  const float* Ab = A + (size_t)b * N_ * N_ + (size_t)mBase * N_;
  const float* hb = h + (size_t)b * N_ * M_;

  // ---- Stage W1^T (bf16) once: Wb[h][m], K = M dim ----
  for (int i = 0; i < 128; ++i) {
    int g  = i * 256 + tid;                   // linear over W1[256][128]
    int m  = g >> 7;
    int hh = g & 127;
    Wb[hh * LDW1 + m] = f2bf(W1[g]);
  }

  // ================= GEMM1: acc = A @ h (fp32 accum) + row-max(A) ==========
  v8f acc[4][4];
  const v8f vzero = {0.f,0.f,0.f,0.f,0.f,0.f,0.f,0.f};
  #pragma unroll
  for (int i = 0; i < 4; ++i)
    #pragma unroll
    for (int j = 0; j < 4; ++j) acc[i][j] = vzero;

  const int mA = tid >> 1;        // staging: fixed row per thread (for row-max)
  const int hk = tid & 1;         // which 16-wide K half this thread covers
  const int kkH  = tid >> 3;      // h-staging row in chunk
  const int grpH = tid & 7;       // h-staging 32-col group
  float amax = -3.402823466e+38f;

  for (int kc = 0; kc < N_ / KC; ++kc) {
    const int kBase = kc * KC;
    // Unconditional speculative prefetch of the next chunk's lines
    // (ISA 10.5: OOB speculative prefetches are silently dropped; no branch,
    //  so the loop body is not peeled/duplicated).
    __builtin_prefetch(Ab + (size_t)mA * N_ + kBase + KC + hk * 16, 0, 0);
    __builtin_prefetch(hb + (size_t)(kBase + KC + kkH) * M_ + grpH * 32, 0, 0);
    __syncthreads();
    // stage A chunk [128 x 32] -> bf16 (packed b64 stores), fold in row max
    {
      const f32x4* src = (const f32x4*)(Ab + (size_t)mA * N_ + kBase + hk * 16);
      u32x2* dst = (u32x2*)(As + mA * LDA + hk * 16);
      #pragma unroll
      for (int j = 0; j < 4; ++j) {
        f32x4 v = src[j];
        amax = fmaxf(amax, fmaxf(fmaxf(v.x, v.y), fmaxf(v.z, v.w)));
        u32x2 pk;
        pk.x = pack_bf16x2(v.x, v.y);
        pk.y = pack_bf16x2(v.z, v.w);
        dst[j] = pk;
      }
    }
    // stage h chunk [32 x 256] transposed -> hT[n][k] bf16
    {
      const f32x4* src = (const f32x4*)(hb + (size_t)(kBase + kkH) * M_ + grpH * 32);
      #pragma unroll
      for (int j4 = 0; j4 < 8; ++j4) {
        f32x4 v = src[j4];
        int n0 = grpH * 32 + j4 * 4;
        hT[(n0+0)*LDH + kkH] = f2bf(v.x);
        hT[(n0+1)*LDH + kkH] = f2bf(v.y);
        hT[(n0+2)*LDH + kkH] = f2bf(v.z);
        hT[(n0+3)*LDH + kkH] = f2bf(v.w);
      }
    }
    __syncthreads();

    Frag af[4], bf[4];
    #pragma unroll
    for (int t = 0; t < 4; ++t) af[t] = load_frag(As, wr*64 + t*16 + r, LDA, 0, hl);
    #pragma unroll
    for (int t = 0; t < 4; ++t) bf[t] = load_frag(hT, wc*64 + t*16 + r, LDH, 0, hl);
    #pragma unroll
    for (int i = 0; i < 4; ++i)
      #pragma unroll
      for (int j = 0; j < 4; ++j)
        acc[i][j] = wmma_bf16(af[i], bf[j], acc[i][j]);
  }

  // ---- row max combine + stage k/3 (bf16) into kb ----
  __syncthreads();
  if (hk == 0) maskLds[mA] = amax;
  __syncthreads();
  if (hk == 1) maskLds[mA] = fmaxf(maskLds[mA], amax);
  const float inv3 = 1.0f / 3.0f;
  #pragma unroll
  for (int i = 0; i < 4; ++i) {
    int mrow = wr*64 + i*16 + hl*8;
    #pragma unroll
    for (int j = 0; j < 4; ++j) {
      int n = wc*64 + j*16 + r;
      #pragma unroll
      for (int e = 0; e < 8; ++e)
        kb[(mrow + e) * LDKB + n] = f2bf(acc[i][j][e] * inv3);
    }
  }
  __syncthreads();

  // ================= GEMM2a: t = relu(kb @ W1 + b1) ========================
  v8f acc2[4][2];
  #pragma unroll
  for (int i = 0; i < 4; ++i) { acc2[i][0] = vzero; acc2[i][1] = vzero; }
  for (int kc = 0; kc < M_ / KC; ++kc) {
    Frag af[4], bw[2];
    #pragma unroll
    for (int t = 0; t < 4; ++t) af[t] = load_frag(kb, wr*64 + t*16 + r, LDKB, kc*32, hl);
    #pragma unroll
    for (int t = 0; t < 2; ++t) bw[t] = load_frag(Wb, wc*32 + t*16 + r, LDW1, kc*32, hl);
    #pragma unroll
    for (int i = 0; i < 4; ++i)
      #pragma unroll
      for (int j = 0; j < 2; ++j)
        acc2[i][j] = wmma_bf16(af[i], bw[j], acc2[i][j]);
  }
  #pragma unroll
  for (int j = 0; j < 2; ++j) {
    int hcol = wc*32 + j*16 + r;
    float bias = b1[hcol];
    #pragma unroll
    for (int i = 0; i < 4; ++i) {
      int mrow = wr*64 + i*16 + hl*8;
      #pragma unroll
      for (int e = 0; e < 8; ++e) {
        float v = acc2[i][j][e] + bias;
        tb[(mrow + e) * LDTB + hcol] = f2bf(v > 0.f ? v : 0.f);
      }
    }
  }
  __syncthreads();

  // ---- restage Wb with W2^T: Wb[n][h], K = H dim ----
  for (int i = 0; i < 128; ++i) {
    int g  = i * 256 + tid;                   // linear over W2[128][256]
    int hh = g >> 8;
    int n  = g & 255;
    Wb[n * LDW2 + hh] = f2bf(W2[g]);
  }
  __syncthreads();

  // ================= GEMM2b + gated epilogue ===============================
  v8f accB[4][4];
  #pragma unroll
  for (int i = 0; i < 4; ++i)
    #pragma unroll
    for (int j = 0; j < 4; ++j) accB[i][j] = vzero;
  for (int kc = 0; kc < H_ / KC; ++kc) {
    Frag af[4], bw[4];
    #pragma unroll
    for (int t = 0; t < 4; ++t) af[t] = load_frag(tb, wr*64 + t*16 + r, LDTB, kc*32, hl);
    #pragma unroll
    for (int t = 0; t < 4; ++t) bw[t] = load_frag(Wb, wc*64 + t*16 + r, LDW2, kc*32, hl);
    #pragma unroll
    for (int i = 0; i < 4; ++i)
      #pragma unroll
      for (int j = 0; j < 4; ++j)
        accB[i][j] = wmma_bf16(af[i], bw[j], accB[i][j]);
  }

  float* outb = out + (size_t)b * N_ * M_ + (size_t)mBase * M_;
  #pragma unroll
  for (int j = 0; j < 4; ++j) {
    int n = wc*64 + j*16 + r;
    float b2v = b2[n];
    #pragma unroll
    for (int i = 0; i < 4; ++i) {
      int mrow = wr*64 + i*16 + hl*8;
      #pragma unroll
      for (int e = 0; e < 8; ++e) {
        int m = mrow + e;
        float mk = maskLds[m];
        float hv = hb[(size_t)(mBase + m) * M_ + n];
        float k2 = accB[i][j][e] + b2v;
        outb[(size_t)m * M_ + n] = k2 * mk + hv * (1.0f - mk);
      }
    }
  }
}

extern "C" void kernel_launch(void* const* d_in, const int* in_sizes, int n_in,
                              void* d_out, int out_size, void* d_ws, size_t ws_size,
                              hipStream_t stream) {
  const float* A  = (const float*)d_in[0];
  const float* h  = (const float*)d_in[1];
  const float* W1 = (const float*)d_in[2];
  const float* b1 = (const float*)d_in[3];
  const float* W2 = (const float*)d_in[4];
  const float* b2 = (const float*)d_in[5];
  float* out = (float*)d_out;

  (void)in_sizes; (void)n_in; (void)out_size; (void)d_ws; (void)ws_size;

  hipFuncSetAttribute((const void*)nodes_fused_kernel,
                      hipFuncAttributeMaxDynamicSharedMemorySize, SMEM_BYTES);

  dim3 grid(B_ * (N_ / BLKR));   // 32 batches * 8 row-slabs = 256 WGs
  nodes_fused_kernel<<<grid, 256, SMEM_BYTES, stream>>>(A, h, W1, b1, W2, b2, out);
}
